// Net_74672301408843
// MI455X (gfx1250) — compile-verified, matrix operand order checked by reading, not run on hardware
//
#include <hip/hip_runtime.h>

// ---------------------------------------------------------------------------
// Fused GNN edge-MLP for MI455X (gfx1250, wave32, WMMA bf16 16x16x32).
//   h = relu(bn1(concat[xi, xj-xi] @ W1^T + b1))   -> folded: [xi|xj] @ W1''^T
//   h = relu(bn2(h @ W2^T + b2))
//   o = h @ W3^T + b3 ; scatter-mean over dst ; tanh
// All intermediates live in LDS; weights pre-folded to bf16 in d_ws.
// Wave mapping: each wave covers BOTH 16-edge M-tiles and 2 (hidden) / 1 (out)
// N-tiles, so every global B-fragment is reused by two WMMAs (halves L2
// weight traffic vs. a 1Mx4N mapping; per-WG weight reads == weight size).
// ---------------------------------------------------------------------------

constexpr int FDIM  = 128;   // node feature dim
constexpr int HD    = 256;   // MLP width (= 2*FDIM)
constexpr int TILE  = 32;    // edges per workgroup (2 WMMA M-tiles)
constexpr int LSTR  = 264;   // padded LDS row stride (bf16 elems): 528B -> conflict-free b128
constexpr float EPSV = 1e-5f;

typedef __attribute__((ext_vector_type(16))) __bf16       bf16x16;
typedef __attribute__((ext_vector_type(8)))  float        floatx8;
typedef __attribute__((ext_vector_type(4)))  float        fltx4;
typedef __attribute__((ext_vector_type(4)))  unsigned int u32x4;
typedef __attribute__((ext_vector_type(2)))  unsigned int u32x2;

union FragAB { u32x4 q[2]; bf16x16 v; };   // 32 bytes: one 16x32 bf16 A/B fragment per lane

__device__ __forceinline__ unsigned short f2bf(float f) {
    union { float f; unsigned int u; } v; v.f = f;
    unsigned int r = v.u + 0x7FFFu + ((v.u >> 16) & 1u);   // round-to-nearest-even
    return (unsigned short)(r >> 16);
}

__device__ __forceinline__ floatx8 wmma_bf16(const FragAB& a, const FragAB& b, floatx8 c) {
    // D = A(16x32 bf16) x B(32x16 bf16) + C(16x16 f32)
    return __builtin_amdgcn_wmma_f32_16x16x32_bf16(false, a.v, false, b.v,
                                                   (short)0, c, false, false);
}

// ---------------------------------------------------------------------------
// Weight prep: fold BN into W/b, fold concat trick into W1, convert to bf16.
// Grid covers exactly HD*HD threads: o = id/256, k = id%256.
// ---------------------------------------------------------------------------
__global__ void prep_kernel(const float* __restrict__ W1, const float* __restrict__ b1,
                            const float* __restrict__ g1, const float* __restrict__ be1,
                            const float* __restrict__ rm1, const float* __restrict__ rv1,
                            const float* __restrict__ W2, const float* __restrict__ b2,
                            const float* __restrict__ g2, const float* __restrict__ be2,
                            const float* __restrict__ rm2, const float* __restrict__ rv2,
                            const float* __restrict__ W3, const float* __restrict__ b3,
                            unsigned short* __restrict__ W1o, unsigned short* __restrict__ W2o,
                            unsigned short* __restrict__ W3o,
                            float* __restrict__ b1o, float* __restrict__ b2o,
                            float* __restrict__ b3o) {
    const int id = blockIdx.x * blockDim.x + threadIdx.x;
    const int o = id >> 8;
    const int k = id & 255;

    const float s1 = g1[o] * rsqrtf(rv1[o] + EPSV);
    float w1 = W1[o * HD + k];
    if (k < FDIM) w1 -= W1[o * HD + FDIM + k];   // concat([xi, xj-xi]) -> [xi, xj]
    W1o[o * HD + k] = f2bf(s1 * w1);

    const float s2 = g2[o] * rsqrtf(rv2[o] + EPSV);
    W2o[o * HD + k] = f2bf(s2 * W2[o * HD + k]);

    if (o < FDIM) W3o[o * HD + k] = f2bf(W3[o * HD + k]);

    if (k == 0) {
        b1o[o] = s1 * (b1[o] - rm1[o]) + be1[o];
        b2o[o] = s2 * (b2[o] - rm2[o]) + be2[o];
        if (o < FDIM) b3o[o] = b3[o];
    }
}

__global__ void zero_kernel(float* __restrict__ p, int n) {
    const int id = blockIdx.x * blockDim.x + threadIdx.x;
    if (id < n) p[id] = 0.f;
}

// ---------------------------------------------------------------------------
// One hidden layer: [TILE x HD] (LDS bf16) x [HD x HD]^T (global bf16) with
// WMMA; bias pre-loaded into accumulators; optional ReLU; bf16 back to LDS.
// The wave computes 2 M-tiles x NT N-tiles; each B-fragment feeds 2 WMMAs.
// ---------------------------------------------------------------------------
template <int NT, bool RELU>
__device__ __forceinline__ void layer_lds(const unsigned short* __restrict__ hin,
                                          unsigned short* __restrict__ hout,
                                          const unsigned short* __restrict__ Wg,
                                          const float* __restrict__ bias,
                                          int nBase, int lane) {
    const int hf  = lane >> 4;           // lane half selects K sub-block
    const int l16 = lane & 15;           // A: row M; B/C: col N
    const unsigned short* aRow0 = hin + l16 * LSTR;          // M-tile 0
    const unsigned short* aRow1 = hin + (16 + l16) * LSTR;   // M-tile 1

    floatx8 acc[2][NT];
#pragma unroll
    for (int j = 0; j < NT; ++j) {        // start from bias (N fixed per lane)
        const float bz = bias[nBase + j * 16 + l16];
        const floatx8 bv = {bz, bz, bz, bz, bz, bz, bz, bz};
        acc[0][j] = bv;
        acc[1][j] = bv;
    }

#pragma unroll
    for (int k0 = 0; k0 < HD; k0 += 32) {
        FragAB a0, a1;                                        // 16x32 bf16 A-fragments
        a0.q[0] = *(const u32x4*)(aRow0 + k0 + hf * 8);       // K = k0 + hf*8  .. +7
        a0.q[1] = *(const u32x4*)(aRow0 + k0 + 16 + hf * 8);  // K = k0+16+hf*8 .. +7
        a1.q[0] = *(const u32x4*)(aRow1 + k0 + hf * 8);
        a1.q[1] = *(const u32x4*)(aRow1 + k0 + 16 + hf * 8);
#pragma unroll
        for (int j = 0; j < NT; ++j) {
            const unsigned short* bRow = Wg + (nBase + j * 16 + l16) * HD;
            FragAB b;                                         // 32x16 bf16 B-fragment
            b.q[0] = *(const u32x4*)(bRow + k0 + hf * 8);
            b.q[1] = *(const u32x4*)(bRow + k0 + 16 + hf * 8);
            acc[0][j] = wmma_bf16(a0, b, acc[0][j]);          // B reused across M-tiles
            acc[1][j] = wmma_bf16(a1, b, acc[1][j]);
        }
    }
#pragma unroll
    for (int mt = 0; mt < 2; ++mt) {
#pragma unroll
        for (int j = 0; j < NT; ++j) {
            const int n = nBase + j * 16 + l16;
#pragma unroll
            for (int r = 0; r < 8; ++r) {                     // C: M = r + 8*hf
                float v = acc[mt][j][r];
                if (RELU) v = fmaxf(v, 0.f);
                hout[(mt * 16 + r + 8 * hf) * LSTR + n] = f2bf(v);
            }
        }
    }
}

// Output layer: [TILE x HD] x [FDIM x HD]^T, bias, atomic scatter-add to accg.
// Wave covers 2 M-tiles x 1 N-tile.
__device__ __forceinline__ void layer_out(const unsigned short* __restrict__ hin,
                                          const unsigned short* __restrict__ Wg,
                                          const float* __restrict__ bias,
                                          float* __restrict__ accg,
                                          const int* __restrict__ sdst,
                                          int nBase, int lane, int eBase, int E) {
    const int hf  = lane >> 4;
    const int l16 = lane & 15;
    const unsigned short* aRow0 = hin + l16 * LSTR;
    const unsigned short* aRow1 = hin + (16 + l16) * LSTR;

    const int n = nBase + l16;
    const float bz = bias[n];
    const floatx8 bv = {bz, bz, bz, bz, bz, bz, bz, bz};
    floatx8 acc[2];
    acc[0] = bv;
    acc[1] = bv;

#pragma unroll
    for (int k0 = 0; k0 < HD; k0 += 32) {
        FragAB a0, a1, b;
        a0.q[0] = *(const u32x4*)(aRow0 + k0 + hf * 8);
        a0.q[1] = *(const u32x4*)(aRow0 + k0 + 16 + hf * 8);
        a1.q[0] = *(const u32x4*)(aRow1 + k0 + hf * 8);
        a1.q[1] = *(const u32x4*)(aRow1 + k0 + 16 + hf * 8);
        const unsigned short* bRow = Wg + n * HD;
        b.q[0] = *(const u32x4*)(bRow + k0 + hf * 8);
        b.q[1] = *(const u32x4*)(bRow + k0 + 16 + hf * 8);
        acc[0] = wmma_bf16(a0, b, acc[0]);
        acc[1] = wmma_bf16(a1, b, acc[1]);
    }
#pragma unroll
    for (int mt = 0; mt < 2; ++mt) {
#pragma unroll
        for (int r = 0; r < 8; ++r) {
            const int m = mt * 16 + r + 8 * hf;
            if (eBase + m < E) {
                atomicAdd(&accg[sdst[m] * FDIM + n], acc[mt][r]);
            }
        }
    }
}

// ---------------------------------------------------------------------------
// Fused edge kernel: gather -> L1 -> L2 -> L3 -> atomic scatter.
// 256 threads = 8 wave32; wave w owns hidden cols [w*32, w*32+32) and
// output cols [w*16, w*16+16), across both M-tiles.
// ---------------------------------------------------------------------------
__global__ __launch_bounds__(256) void edge_kernel(
    const float* __restrict__ x,
    const int* __restrict__ srcIdx, const int* __restrict__ dstIdx,
    const unsigned short* __restrict__ W1o, const unsigned short* __restrict__ W2o,
    const unsigned short* __restrict__ W3o,
    const float* __restrict__ b1o, const float* __restrict__ b2o,
    const float* __restrict__ b3o,
    float* __restrict__ accg, float* __restrict__ cnt, int E) {

    __shared__ unsigned short hA[TILE * LSTR];   // 16.9 KB: [xi|xj] bf16, then h2
    __shared__ unsigned short hB[TILE * LSTR];   // 16.9 KB: h1
    __shared__ int sdst[TILE];

    const int tid  = threadIdx.x;
    const int base = blockIdx.x * TILE;

    if (tid < TILE) {
        const int e  = base + tid;
        const int ec = e < E ? e : E - 1;
        const int d  = dstIdx[ec];
        sdst[tid] = d;
        if (e < E) atomicAdd(&cnt[d], 1.0f);     // degree for scatter-mean
    }
    __syncthreads();

    // Gather x[dst] into cols [0,128) and x[src] into cols [128,256) as bf16.
    // 32 edges x 2 halves x 32 float4-chunks = 2048 tasks, chunk fastest (coalesced).
    for (int t = tid; t < TILE * 2 * 32; t += 256) {
        const int c    = t & 31;
        const int rest = t >> 5;
        const int sh   = rest & 1;               // 0 = xi (dst), 1 = xj (src)
        const int row  = rest >> 1;
        const int e    = base + row;
        const int ec   = e < E ? e : E - 1;
        const int node = sh ? srcIdx[ec] : sdst[row];
        const fltx4 vx = *(const fltx4*)(x + node * FDIM + c * 4);
        u32x2 pk;
        pk.x = (unsigned)f2bf(vx.x) | ((unsigned)f2bf(vx.y) << 16);
        pk.y = (unsigned)f2bf(vx.z) | ((unsigned)f2bf(vx.w) << 16);
        *(u32x2*)(hA + row * LSTR + sh * FDIM + c * 4) = pk;
    }
    __syncthreads();

    const int lane = tid & 31;
    const int wave = tid >> 5;

    layer_lds<2, true>(hA, hB, W1o, b1o, wave * 32, lane);   // 256-wide, ReLU
    __syncthreads();
    layer_lds<2, true>(hB, hA, W2o, b2o, wave * 32, lane);   // 256-wide, ReLU
    __syncthreads();
    layer_out(hA, W3o, b3o, accg, sdst, wave * 16, lane, base, E);  // 128-wide
}

__global__ void finish_kernel(const float* __restrict__ accg,
                              const float* __restrict__ cnt,
                              float* __restrict__ out, int total) {
    const int id = blockIdx.x * blockDim.x + threadIdx.x;
    if (id >= total) return;
    const int node = id >> 7;                    // / FDIM
    const float c  = cnt[node];
    out[id] = tanhf(accg[id] / fmaxf(c, 1.0f));
}

// ---------------------------------------------------------------------------
extern "C" void kernel_launch(void* const* d_in, const int* in_sizes, int n_in,
                              void* d_out, int out_size, void* d_ws, size_t ws_size,
                              hipStream_t stream) {
    (void)n_in; (void)out_size; (void)ws_size;

    const float* x   = (const float*)d_in[0];
    const int*   ei  = (const int*)d_in[1];     // harness: integer -> int32; [2][E]
    const float* W1  = (const float*)d_in[2];
    const float* b1  = (const float*)d_in[3];
    const float* g1  = (const float*)d_in[4];
    const float* be1 = (const float*)d_in[5];
    const float* rm1 = (const float*)d_in[6];
    const float* rv1 = (const float*)d_in[7];
    const float* W2  = (const float*)d_in[8];
    const float* b2  = (const float*)d_in[9];
    const float* g2  = (const float*)d_in[10];
    const float* be2 = (const float*)d_in[11];
    const float* rm2 = (const float*)d_in[12];
    const float* rv2 = (const float*)d_in[13];
    const float* W3  = (const float*)d_in[14];
    const float* b3  = (const float*)d_in[15];

    const int E = in_sizes[1] / 2;
    const int N = in_sizes[0] / FDIM;
    const int* srcIdx = ei;          // edge_index[0]
    const int* dstIdx = ei + E;      // edge_index[1]

    // Workspace layout (bytes)
    char* ws = (char*)d_ws;
    unsigned short* W1o = (unsigned short*)(ws + 0);         // 256x256 bf16 = 128 KB
    unsigned short* W2o = (unsigned short*)(ws + 131072);    // 128 KB
    unsigned short* W3o = (unsigned short*)(ws + 262144);    // 128x256 bf16 = 64 KB
    float* b1o  = (float*)(ws + 327680);
    float* b2o  = (float*)(ws + 328704);
    float* b3o  = (float*)(ws + 329728);
    float* accg = (float*)(ws + 330240);                     // N*FDIM f32
    float* cnt  = accg + (size_t)N * FDIM;                   // N f32 (contiguous)

    float* out = (float*)d_out;

    prep_kernel<<<(HD * HD) / 256, 256, 0, stream>>>(
        W1, b1, g1, be1, rm1, rv1, W2, b2, g2, be2, rm2, rv2, W3, b3,
        W1o, W2o, W3o, b1o, b2o, b3o);

    const int zn = N * FDIM + N;
    zero_kernel<<<(zn + 255) / 256, 256, 0, stream>>>(accg, zn);

    edge_kernel<<<(E + TILE - 1) / TILE, 256, 0, stream>>>(
        x, srcIdx, dstIdx, W1o, W2o, W3o, b1o, b2o, b3o, accg, cnt, E);

    finish_kernel<<<(N * FDIM + 255) / 256, 256, 0, stream>>>(accg, cnt, out, N * FDIM);
}
